// SeqGraph_18863496364475
// MI455X (gfx1250) — matrix operand order repaired and checked by reading, not compiled, not run on hardware
//
#include <hip/hip_runtime.h>
#include <math.h>

#define DIM      128
#define HID_NUM  16
#define HID_SZ   10
#define HS       160      // HID_NUM*HID_SZ
#define NNODES   100000
#define NEDGES   1600000
#define NGRAPHS  1024
#define PAIRS    45       // HID_SZ*(HID_SZ-1)/2
#define OUTW     64       // HID_NUM*MAX_STEP

typedef __attribute__((ext_vector_type(16))) __bf16 v16bf;
typedef __attribute__((ext_vector_type(8)))  __bf16 v8bf;
typedef __attribute__((ext_vector_type(8)))  float  v8f;

static __device__ __forceinline__ __bf16 f2bf(float f) {
  union { float f; unsigned u; } v; v.f = f;
  unsigned r = (v.u + 0x7FFFu + ((v.u >> 16) & 1u)) >> 16;  // round-to-nearest-even
  unsigned short s = (unsigned short)r;
  __bf16 o; __builtin_memcpy(&o, &s, 2);
  return o;
}

static __device__ __forceinline__ v16bf cat8(v8bf lo, v8bf hi) {
  return __builtin_shufflevector(lo, hi, 0, 1, 2, 3, 4, 5, 6, 7,
                                         8, 9, 10, 11, 12, 13, 14, 15);
}

static __device__ __forceinline__ float lrelu(float v) { return v > 0.f ? v : 0.01f * v; }

// ---- dense symmetric hidden-graph adjacency from packed upper triangle ----
__global__ void build_A_kernel(const float* __restrict__ hid_adj, float* __restrict__ A) {
  int idx = blockIdx.x * blockDim.x + threadIdx.x;
  if (idx >= HID_NUM * PAIRS) return;
  int h = idx / PAIRS, p = idx % PAIRS;
  int i = 0, off = p;
  while (off >= (HID_SZ - 1 - i)) { off -= (HID_SZ - 1 - i); ++i; }
  int j = i + 1 + off;
  float v = lrelu(hid_adj[idx]);
  A[h * 100 + i * 10 + j] = v;
  A[h * 100 + j * 10 + i] = v;
}

// ---- fc_w [k][col] -> bf16 transposed Wt[col][k] (done once, 32 KB, L2-resident) ----
__global__ void cvt_wt_kernel(const float* __restrict__ w, __bf16* __restrict__ wt) {
  int idx = blockIdx.x * 256 + threadIdx.x;     // 16384
  int k = idx >> 7, col = idx & 127;
  wt[col * DIM + k] = f2bf(w[k * DIM + col]);
}

// ---- generic f32 -> bf16 convert (used for hid_feat) ----
__global__ void cvt_bf_kernel(const float* __restrict__ in, __bf16* __restrict__ out, int n) {
  int idx = blockIdx.x * 256 + threadIdx.x;
  if (idx < n) out[idx] = f2bf(in[idx]);
}

// ---- x0 = sigmoid(poi_table[node_x] @ fc_w + fc_b) via bf16 WMMA ----
// block: 8 waves, each wave one 16-col tile; gathered rows staged once in LDS as bf16.
__global__ __launch_bounds__(256) void fc_kernel(
    const int* __restrict__ node_x, const float* __restrict__ poi_table,
    const __bf16* __restrict__ wtbf, const float* __restrict__ fc_b,
    float* __restrict__ x_out) {
  __shared__ alignas(32) __bf16 atile[16 * DIM];   // gathered+converted embedding rows
  const int ntile = blockIdx.x;                    // node tile 0..6249
  for (int i = threadIdx.x; i < 16 * DIM; i += 256) {
    int r = i >> 7, k = i & 127;
    atile[i] = f2bf(poi_table[(long)node_x[ntile * 16 + r] * DIM + k]);
  }
  __syncthreads();

  const int lane = threadIdx.x & 31;
  const int wave = threadIdx.x >> 5;   // column tile 0..7
  const int hi   = lane >> 4;
  const int mloc = lane & 15;
  const int col  = wave * 16 + mloc;

  v8f c = {};
  #pragma unroll
  for (int kc = 0; kc < DIM; kc += 32) {
    const int ka = kc + hi * 16;       // A: K 0-15 (lo half), 16-31 (hi half)
    v8bf alo = *(const v8bf*)&atile[mloc * DIM + ka];
    v8bf ahi = *(const v8bf*)&atile[mloc * DIM + ka + 8];
    v16bf a = cat8(alo, ahi);
    const int kb = kc + hi * 16;       // B: K 0-15 (lo half), 16-31 (hi half)
    v16bf b = *(const v16bf*)&wtbf[col * DIM + kb];
    c = __builtin_amdgcn_wmma_f32_16x16x32_bf16(false, a, false, b, (short)0, c, false, false);
  }
  const float bias = fc_b[col];
  #pragma unroll
  for (int r = 0; r < 8; ++r) {
    int row = ntile * 16 + r + hi * 8;
    float e = __expf(-(c[r] + bias));
    x_out[(long)row * DIM + col] = __builtin_amdgcn_rcpf(1.f + e);   // sigmoid
  }
}

// ---- fused: T = z_i · x_i  (WMMA),  t = ZX*T, Σ_s, segment-sum into out_g ----
// block: 10 waves (one per 16-row hs tile), one 16-node tile; x tile staged bf16 in LDS.
__global__ __launch_bounds__(320) void einsum_pool_kernel(
    const __bf16* __restrict__ zbf,  // [160][128] bf16
    const float* __restrict__ x,     // [N][128] f32
    float* __restrict__ zx,          // [N][160]: stored at step 0, read after
    const int* __restrict__ batch,
    float* __restrict__ out_g,       // [G][64]
    int step) {
  __shared__ alignas(32) __bf16 xtile[16 * DIM];
  __shared__ float acc[16 * 16];     // [node_in_tile][h]
  const int ntile = blockIdx.x;
  const int t = threadIdx.x;
  for (int i = t; i < 16 * DIM; i += 320)
    xtile[i] = f2bf(x[(long)ntile * 16 * DIM + i]);
  if (t < 256) acc[t] = 0.f;
  __syncthreads();

  const int lane = t & 31;
  const int wave = t >> 5;           // hs tile 0..9
  const int hi   = lane >> 4;
  const int nloc = lane & 15;
  const int hsrow = wave * 16 + nloc;

  v8f c = {};
  #pragma unroll
  for (int kc = 0; kc < DIM; kc += 32) {
    const int ka = kc + hi * 16;
    v8bf alo = *(const v8bf*)&zbf[hsrow * DIM + ka];
    v8bf ahi = *(const v8bf*)&zbf[hsrow * DIM + ka + 8];
    v16bf a = cat8(alo, ahi);
    const int kb = kc + hi * 16;
    v16bf b = *(const v16bf*)&xtile[nloc * DIM + kb];
    c = __builtin_amdgcn_wmma_f32_16x16x32_bf16(false, a, false, b, (short)0, c, false, false);
  }

  const int node = ntile * 16 + nloc;
  #pragma unroll
  for (int r = 0; r < 8; ++r) {
    int hs = wave * 16 + r + hi * 8;
    float tv = c[r], zxv;
    if (step == 0) { zx[(long)node * HS + hs] = tv; zxv = tv; }
    else           { zxv = zx[(long)node * HS + hs]; }
    atomicAdd(&acc[nloc * 16 + (hs / HID_SZ)], tv * zxv);   // ds_add_f32
  }
  __syncthreads();
  if (t < 256) {
    int ni = t >> 4, h = t & 15;
    int g = batch[ntile * 16 + ni];
    atomicAdd(&out_g[g * OUTW + step * HID_NUM + h], acc[ni * 16 + h]);
  }
}

// ---- x'[dst] += x[src] over all edges: one wave per edge, float4 per lane ----
// x buffers (51 MB) are L2-resident (192 MB L2) -> atomics resolve at L2 bandwidth.
__global__ __launch_bounds__(256) void propagate_kernel(
    const int* __restrict__ edge_index, const float* __restrict__ x_in,
    float* __restrict__ x_out) {
  const long gid = (long)blockIdx.x * 256 + threadIdx.x;
  const int e = (int)(gid >> 5);           // uniform per wave
  const int lane = threadIdx.x & 31;
  const int src = edge_index[e];
  const int dst = edge_index[NEDGES + e];
  const float4 v = *reinterpret_cast<const float4*>(x_in + (long)src * DIM + lane * 4);
  float* o = x_out + (long)dst * DIM + lane * 4;
  atomicAdd(o + 0, v.x);
  atomicAdd(o + 1, v.y);
  atomicAdd(o + 2, v.z);
  atomicAdd(o + 3, v.w);
}

// ---- z' = A @ z per hidden graph (tiny); also emits bf16 copy for the WMMA kernel ----
__global__ void zmul_kernel(const float* __restrict__ A, const float* __restrict__ zin,
                            float* __restrict__ zout, __bf16* __restrict__ zbf) {
  const int h = blockIdx.x, d = threadIdx.x;
  #pragma unroll
  for (int s = 0; s < HID_SZ; ++s) {
    float acc = 0.f;
    #pragma unroll
    for (int tt = 0; tt < HID_SZ; ++tt)
      acc += A[h * 100 + s * 10 + tt] * zin[(h * HID_SZ + tt) * DIM + d];
    zout[(h * HID_SZ + s) * DIM + d] = acc;
    zbf[(h * HID_SZ + s) * DIM + d] = f2bf(acc);
  }
}

// ---- head MLPs (tiny; plain VALU) ----
__global__ void sess_kernel(const float* __restrict__ out_g, const float* __restrict__ mlp_w,
                            const float* __restrict__ mlp_b, float* __restrict__ sess) {
  const int idx = blockIdx.x * 256 + threadIdx.x;
  const int g = idx >> 7, d = idx & 127;
  float a = mlp_b[d];
  for (int k = 0; k < OUTW; ++k) a += out_g[g * OUTW + k] * mlp_w[k * DIM + d];
  sess[idx] = lrelu(a);
}

__global__ void hidden_kernel(const float* __restrict__ sess,
                              const float* __restrict__ proj1_w, const float* __restrict__ proj1_b,
                              const float* __restrict__ pred1_w, const float* __restrict__ pred1_b,
                              const int* __restrict__ tar_poi, const float* __restrict__ poi_table,
                              float* __restrict__ p1, float* __restrict__ h1) {
  const int idx = blockIdx.x * 256 + threadIdx.x;
  const int g = idx >> 7, d = idx & 127;
  float a = proj1_b[d], b = pred1_b[d];
  for (int k = 0; k < DIM; ++k) {
    float sv = sess[g * DIM + k];
    a += sv * proj1_w[k * DIM + d];
    b += sv * pred1_w[k * DIM + d];
  }
  const float* te = poi_table + (long)tar_poi[g] * DIM;
  for (int k = 0; k < DIM; ++k) b += te[k] * pred1_w[(DIM + k) * DIM + d];
  p1[idx] = lrelu(a);
  h1[idx] = lrelu(b);
}

__global__ void out_kernel(const float* __restrict__ p1,
                           const float* __restrict__ proj2_w, const float* __restrict__ proj2_b,
                           const float* __restrict__ h1,
                           const float* __restrict__ pred2_w, const float* __restrict__ pred2_b,
                           float* __restrict__ out) {
  const int idx = blockIdx.x * 256 + threadIdx.x;
  const int g = idx >> 7, d = idx & 127;
  float a = proj2_b[d];
  for (int k = 0; k < DIM; ++k) a += p1[g * DIM + k] * proj2_w[k * DIM + d];
  out[idx] = a;                               // proj [G,128]
  if (d == 0) {
    float l = pred2_b[0];
    for (int k = 0; k < DIM; ++k) l += h1[g * DIM + k] * pred2_w[k];
    out[NGRAPHS * DIM + g] = l;               // logits [G]
  }
}

extern "C" void kernel_launch(void* const* d_in, const int* in_sizes, int n_in,
                              void* d_out, int out_size, void* d_ws, size_t ws_size,
                              hipStream_t stream) {
  const int*   node_x    = (const int*)d_in[0];
  const int*   edge_idx  = (const int*)d_in[1];
  const int*   batch     = (const int*)d_in[2];
  const int*   tar_poi   = (const int*)d_in[3];
  const float* poi_table = (const float*)d_in[5];
  const float* fc_w      = (const float*)d_in[6];
  const float* fc_b      = (const float*)d_in[7];
  const float* hid_adj   = (const float*)d_in[8];
  const float* hid_feat  = (const float*)d_in[9];
  const float* mlp_w     = (const float*)d_in[10];
  const float* mlp_b     = (const float*)d_in[11];
  const float* proj1_w   = (const float*)d_in[12];
  const float* proj1_b   = (const float*)d_in[13];
  const float* proj2_w   = (const float*)d_in[14];
  const float* proj2_b   = (const float*)d_in[15];
  const float* pred1_w   = (const float*)d_in[16];
  const float* pred1_b   = (const float*)d_in[17];
  const float* pred2_w   = (const float*)d_in[18];
  const float* pred2_b   = (const float*)d_in[19];

  float* ws = (float*)d_ws;
  size_t o = 0;
  float* xA    = ws + o; o += (size_t)NNODES * DIM;
  float* xB    = ws + o; o += (size_t)NNODES * DIM;
  float* zxbuf = ws + o; o += (size_t)NNODES * HS;
  float* A     = ws + o; o += HID_NUM * 100;
  float* zA    = ws + o; o += HS * DIM;
  float* zB    = ws + o; o += HS * DIM;
  float* outg  = ws + o; o += (size_t)NGRAPHS * OUTW;
  float* sess  = ws + o; o += (size_t)NGRAPHS * DIM;
  float* p1    = ws + o; o += (size_t)NGRAPHS * DIM;
  float* h1    = ws + o; o += (size_t)NGRAPHS * DIM;
  __bf16* wtbf = (__bf16*)(ws + o); o += DIM * DIM / 2;   // 128x128 bf16
  __bf16* zbfA = (__bf16*)(ws + o); o += HS * DIM / 2;    // 160x128 bf16
  __bf16* zbfB = (__bf16*)(ws + o); o += HS * DIM / 2;

  const int NT   = NNODES / 16;                     // 6250 node tiles
  const int EGRD = (int)((long)NEDGES * 32 / 256);  // 200000

  hipMemsetAsync(A, 0, HID_NUM * 100 * sizeof(float), stream);
  hipMemsetAsync(outg, 0, (size_t)NGRAPHS * OUTW * sizeof(float), stream);
  build_A_kernel<<<(HID_NUM * PAIRS + 255) / 256, 256, 0, stream>>>(hid_adj, A);
  cvt_wt_kernel<<<DIM * DIM / 256, 256, 0, stream>>>(fc_w, wtbf);
  cvt_bf_kernel<<<(HS * DIM + 255) / 256, 256, 0, stream>>>(hid_feat, zbfB, HS * DIM);

  fc_kernel<<<NT, 256, 0, stream>>>(node_x, poi_table, wtbf, fc_b, xA);

  // step 0: z = hid_feat (bf16 in zbfB), x = x0; stores ZX and pools
  einsum_pool_kernel<<<NT, 320, 0, stream>>>(zbfB, xA, zxbuf, batch, outg, 0);

  // step 1
  hipMemsetAsync(xB, 0, (size_t)NNODES * DIM * sizeof(float), stream);
  propagate_kernel<<<EGRD, 256, 0, stream>>>(edge_idx, xA, xB);
  zmul_kernel<<<HID_NUM, DIM, 0, stream>>>(A, hid_feat, zA, zbfA);
  einsum_pool_kernel<<<NT, 320, 0, stream>>>(zbfA, xB, zxbuf, batch, outg, 1);

  // step 2
  hipMemsetAsync(xA, 0, (size_t)NNODES * DIM * sizeof(float), stream);
  propagate_kernel<<<EGRD, 256, 0, stream>>>(edge_idx, xB, xA);
  zmul_kernel<<<HID_NUM, DIM, 0, stream>>>(A, zA, zB, zbfB);
  einsum_pool_kernel<<<NT, 320, 0, stream>>>(zbfB, xA, zxbuf, batch, outg, 2);

  // step 3
  hipMemsetAsync(xB, 0, (size_t)NNODES * DIM * sizeof(float), stream);
  propagate_kernel<<<EGRD, 256, 0, stream>>>(edge_idx, xA, xB);
  zmul_kernel<<<HID_NUM, DIM, 0, stream>>>(A, zB, zA, zbfA);
  einsum_pool_kernel<<<NT, 320, 0, stream>>>(zbfA, xB, zxbuf, batch, outg, 3);

  // head
  const int HG = NGRAPHS * DIM / 256;               // 512
  sess_kernel<<<HG, 256, 0, stream>>>(outg, mlp_w, mlp_b, sess);
  hidden_kernel<<<HG, 256, 0, stream>>>(sess, proj1_w, proj1_b, pred1_w, pred1_b,
                                        tar_poi, poi_table, p1, h1);
  out_kernel<<<HG, 256, 0, stream>>>(p1, proj2_w, proj2_b, h1, pred2_w, pred2_b,
                                     (float*)d_out);
}
